// MoE_34943853920559
// MI455X (gfx1250) — matrix operand order repaired and checked by reading, not compiled, not run on hardware
//
#include <hip/hip_runtime.h>
#include <hip/hip_bf16.h>
#include <math.h>

// Problem constants (B,T,D,H,E,K) = (4,1024,1024,4096,8,2)
#define BT   4096   // B*T tokens
#define DD   1024   // model dim
#define HH   4096   // hidden dim
#define EE   8      // experts
#define HC   128    // hidden chunk per iteration
#define NCHUNK (HH / HC)
#define MT   16     // token tile (WMMA M)
#define WSLAB_STRIDE 129   // 128 f32 + 1 dword TDM pad per row
#define HS_STRIDE (HC + 8)

typedef __bf16 bf16;
typedef bf16  v16bf __attribute__((ext_vector_type(16)));
typedef float v8f   __attribute__((ext_vector_type(8)));
typedef unsigned int u32;
typedef u32  u32x4 __attribute__((ext_vector_type(4)));
typedef int  i32x4 __attribute__((ext_vector_type(4)));
typedef int  i32x8 __attribute__((ext_vector_type(8)));

// ---------------------------------------------------------------------------
// Kernel 0: zero output + expert counters
// ---------------------------------------------------------------------------
__global__ void moe_zero_kernel(float* __restrict__ out, int n, int* __restrict__ cnt) {
    int i = blockIdx.x * blockDim.x + threadIdx.x;
    int stride = gridDim.x * blockDim.x;
    for (; i < n; i += stride) out[i] = 0.0f;
    if (blockIdx.x == 0 && threadIdx.x < EE) cnt[threadIdx.x] = 0;
}

// ---------------------------------------------------------------------------
// Kernel 1: router (x @ Wr + br), top-2, softmax, build per-expert lists.
// One wave32 per token.
// ---------------------------------------------------------------------------
__global__ void moe_router_kernel(const float* __restrict__ x,
                                  const float* __restrict__ Wr,
                                  const float* __restrict__ br,
                                  float* __restrict__ probs,
                                  int* __restrict__ cnt,
                                  int* __restrict__ list) {
    const int tok  = (blockIdx.x * blockDim.x + threadIdx.x) >> 5;
    const int lane = threadIdx.x & 31;
    if (tok >= BT) return;

    const float* xr = x + (size_t)tok * DD;
    float s[EE];
#pragma unroll
    for (int e = 0; e < EE; ++e) s[e] = 0.0f;

    for (int d = lane; d < DD; d += 32) {
        const float xv = xr[d];
        const float* wrow = Wr + (size_t)d * EE;
#pragma unroll
        for (int e = 0; e < EE; ++e) s[e] += xv * wrow[e];
    }
#pragma unroll
    for (int e = 0; e < EE; ++e) {
#pragma unroll
        for (int off = 16; off > 0; off >>= 1)
            s[e] += __shfl_xor(s[e], off, 32);
    }

    if (lane == 0) {
#pragma unroll
        for (int e = 0; e < EE; ++e) s[e] += br[e];
        int i0 = 0; float b0 = s[0];
        for (int e = 1; e < EE; ++e) if (s[e] > b0) { b0 = s[e]; i0 = e; }
        int i1 = -1; float b1 = -3.402823466e38f;
        for (int e = 0; e < EE; ++e) if (e != i0 && s[e] > b1) { b1 = s[e]; i1 = e; }
        const float m  = fmaxf(b0, b1);
        const float e0 = __expf(b0 - m), e1 = __expf(b1 - m);
        const float inv = 1.0f / (e0 + e1);
        probs[tok * 2 + 0] = e0 * inv;
        probs[tok * 2 + 1] = e1 * inv;
        int p0 = atomicAdd(&cnt[i0], 1); list[i0 * BT + p0] = tok * 2 + 0;
        int p1 = atomicAdd(&cnt[i1], 1); list[i1 * BT + p1] = tok * 2 + 1;
    }
}

// Branchless tanh-GELU: x * sigmoid(1.595769f*x + 0.071354f*x^3)
__device__ __forceinline__ float gelu_fast(float v) {
    const float u = 1.5957691216057308f * v + 0.07135481282161965f * v * v * v;
    return v / (1.0f + __expf(-u));
}

// ---------------------------------------------------------------------------
// Kernel 2: fused gathered expert MLP.
// grid = (BT/MT, EE); block = 256 threads = 8 waves.
//   per H-chunk of 128:
//     W1 K-slab [32 x 128] f32 staged to LDS by the Tensor Data Mover
//     h[16,128]  = x_tile @ W1[:,chunk]   (bf16 WMMA, f32 acc)
//     h          = gelu(h + b1) -> LDS (overlaid on slab; disjoint lifetime)
//     out[16,1024] += h @ W2[chunk,:]     (bf16 WMMA, f32 acc)
// LDS ~50 KB -> 6 blocks per 320 KB WGP for TDM/VMEM latency hiding.
// ---------------------------------------------------------------------------
__global__ __launch_bounds__(256)
void moe_mlp_kernel(const float* __restrict__ x,
                    const float* __restrict__ W1,
                    const float* __restrict__ b1,
                    const float* __restrict__ W2,
                    const float* __restrict__ b2,
                    const float* __restrict__ probs,
                    const int* __restrict__ cnt,
                    const int* __restrict__ list,
                    float* __restrict__ out) {
    __shared__ bf16  xs[MT][DD + 16];             // 33.3 KB, padded stride
    __shared__ float wslab[32 * WSLAB_STRIDE];    // 16.5 KB: TDM dest / GELU buffer
    __shared__ float sprob[MT];
    __shared__ int   stok[MT];
    bf16* const hsb = (bf16*)wslab;               // overlaid h chunk [MT][HS_STRIDE]

    const int e    = blockIdx.y;
    const int tile = blockIdx.x;
    const int n    = cnt[e];
    if (tile * MT >= n) return;          // uniform exit: EXEC all-ones below

    const int tid  = threadIdx.x;
    const int wave = tid >> 5;
    const int lane = tid & 31;
    const int hi   = lane >> 4;
    const int ln   = lane & 15;

    if (tid < MT) {
        const int idx = tile * MT + tid;
        if (idx < n) {
            const int a = list[e * BT + idx];
            stok[tid]  = a >> 1;
            sprob[tid] = probs[a];
        } else { stok[tid] = 0; sprob[tid] = 0.0f; }
    }
    __syncthreads();

    // Stage gathered x rows into LDS as bf16
    {
        const int row = tid >> 4;
        const int cb  = (tid & 15) * 64;
        const float* src = x + (size_t)stok[row] * DD;
        const bool act = (sprob[row] != 0.0f);
        for (int c = 0; c < 64; c += 4) {
            float4 v = act ? *(const float4*)(src + cb + c) : make_float4(0.f, 0.f, 0.f, 0.f);
            xs[row][cb + c + 0] = (bf16)v.x;
            xs[row][cb + c + 1] = (bf16)v.y;
            xs[row][cb + c + 2] = (bf16)v.z;
            xs[row][cb + c + 3] = (bf16)v.w;
        }
    }

    const float* W1e = W1 + (size_t)e * DD * HH;   // [D, H] row-major
    const float* W2e = W2 + (size_t)e * HH * DD;   // [H, D] row-major
    const u32 slab_lds = (u32)(uintptr_t)(&wslab[0]);   // LDS byte offset

    v8f acc[8];
#pragma unroll
    for (int t = 0; t < 8; ++t) acc[t] = (v8f)(0.0f);

    for (int hc = 0; hc < NCHUNK; ++hc) {
        const int hbase = hc * HC;
        const int nb    = wave * 16;

        // ---- GEMM1 over K = D, W1 K-slabs staged by TDM ----
        v8f hacc = (v8f)(0.0f);
        for (int k0 = 0; k0 < DD; k0 += 32) {
            __syncthreads();   // slab/hs region free for overwrite
            if (wave == 0) {
#if __has_builtin(__builtin_amdgcn_tensor_load_to_lds)
                // D# per CDNA5 ISA ch.8: 2D tile [32 rows x 128 cols] f32,
                // row stride = H, LDS pad 1 dword per 128-dword row.
                const unsigned long long ga =
                    (unsigned long long)(uintptr_t)(W1e + (size_t)k0 * HH + hbase);
                u32x4 g0;
                g0[0] = 1u;                                   // count=1, no gather
                g0[1] = slab_lds;                             // lds_addr
                g0[2] = (u32)(ga & 0xFFFFFFFFu);              // global_addr[31:0]
                g0[3] = (u32)((ga >> 32) & 0x1FFFFFFu) | (2u << 30);  // [56:32] | type=2
                i32x8 g1;
                g1[0] = (int)((2u << 16) | (1u << 20) | (6u << 22));  // 4B elems, pad 1dw/128dw
                g1[1] = (int)0xFFFF0000u;                     // tensor_dim0 lo16 (large)
                g1[2] = (int)(0x7FFFu | (0xFFFFu << 16));     // tensor_dim0 hi / dim1 lo
                g1[3] = (int)(0x7FFFu | ((u32)HC << 16));     // tensor_dim1 hi / tile_dim0=128
                g1[4] = 32;                                   // tile_dim1=32, tile_dim2=0
                g1[5] = HH;                                   // tensor_dim0_stride lo32
                g1[6] = 0;
                g1[7] = 0;
                i32x4 z4 = (i32x4)0;
#if defined(__clang_major__) && __clang_major__ >= 23
                i32x8 z8 = (i32x8)0;
                __builtin_amdgcn_tensor_load_to_lds(g0, g1, z4, z4, z8, 0);
#else
                __builtin_amdgcn_tensor_load_to_lds(g0, g1, z4, z4, 0);
#endif
                __builtin_amdgcn_s_wait_tensorcnt(0);
#else
                // Fallback: cooperative load (wave 0 only; 32 rows x 128 cols)
                for (int i = lane; i < 32 * 32; i += 32) {
                    const int r = i >> 5, c4 = (i & 31) * 4;
                    float4 v = *(const float4*)(W1e + (size_t)(k0 + r) * HH + hbase + c4);
                    float* d = &wslab[r * WSLAB_STRIDE + c4];
                    d[0] = v.x; d[1] = v.y; d[2] = v.z; d[3] = v.w;
                }
#endif
            }
            __syncthreads();   // slab visible to all waves

            // A (16x32 bf16) from xs: lane<16 -> K {0..15}; lane>=16 -> {16..31}
            union { v16bf v; uint4 q[2]; } A;
            const bf16* ap = &xs[ln][k0 + hi * 16];
            A.q[0] = *(const uint4*)ap;
            A.q[1] = *(const uint4*)(ap + 8);
            // B (32x16) from LDS slab: VGPR j = rows 2j,2j+1; half-waves K 0-15/16-31
            union { v16bf v; } Bv;
            const float* sl = &wslab[(hi * 16) * WSLAB_STRIDE + nb + ln];
#pragma unroll
            for (int j = 0; j < 16; ++j) Bv.v[j] = (bf16)sl[j * WSLAB_STRIDE];
            hacc = __builtin_amdgcn_wmma_f32_16x16x32_bf16(
                false, A.v, false, Bv.v, (short)0, hacc, false, false);
        }
        __syncthreads();   // all slab reads done before h chunk overwrites region

        // bias + branchless GELU, pack h chunk into overlaid LDS buffer
        {
            const float b1c = b1[(size_t)e * HH + hbase + nb + ln];
#pragma unroll
            for (int r = 0; r < 8; ++r) {
                const int m = r + hi * 8;
                hsb[m * HS_STRIDE + nb + ln] = (bf16)gelu_fast(hacc[r] + b1c);
            }
        }
        __syncthreads();   // h chunk visible

        // ---- GEMM2: acc[16 x 128] += h[16 x 128] @ W2[chunk, wave cols] ----
        for (int kc = 0; kc < HC; kc += 32) {
            union { v16bf v; uint4 q[2]; } A;
            const bf16* ap = &hsb[ln * HS_STRIDE + kc + hi * 16];
            A.q[0] = *(const uint4*)ap;
            A.q[1] = *(const uint4*)(ap + 8);
            const int kb = hbase + kc + hi * 16;
            // prefetch next K-slab of W2 into cache
            if (kb + 32 < HH)
                __builtin_prefetch(W2e + (size_t)(kb + 32) * DD + wave * 128 + ln * 8, 0, 1);
#pragma unroll
            for (int t = 0; t < 8; ++t) {
                const int col = wave * 128 + t * 16 + ln;
                union { v16bf v; } Bv;
                const float* bp = W2e + (size_t)kb * DD + col;
#pragma unroll
                for (int j = 0; j < 16; ++j) Bv.v[j] = (bf16)bp[(size_t)j * DD];
                acc[t] = __builtin_amdgcn_wmma_f32_16x16x32_bf16(
                    false, A.v, false, Bv.v, (short)0, acc[t], false, false);
            }
        }
        // no trailing barrier: next chunk's k-loop-top barrier fences the
        // overlaid region before the TDM overwrites it
    }

    // ---- epilogue: + b2, scale by routing prob, combine into d_out ----
#pragma unroll
    for (int t = 0; t < 8; ++t) {
        const int col = wave * 128 + t * 16 + ln;
        const float b2c = b2[(size_t)e * DD + col];
#pragma unroll
        for (int r = 0; r < 8; ++r) {
            const int m = r + hi * 8;
            const float p = sprob[m];
            if (p != 0.0f) {
                atomicAdd(&out[(size_t)stok[m] * DD + col], p * (acc[t][r] + b2c));
            }
        }
    }
}

// ---------------------------------------------------------------------------
extern "C" void kernel_launch(void* const* d_in, const int* in_sizes, int n_in,
                              void* d_out, int out_size, void* d_ws, size_t ws_size,
                              hipStream_t stream) {
    const float* x  = (const float*)d_in[0];
    const float* Wr = (const float*)d_in[1];
    const float* br = (const float*)d_in[2];
    const float* W1 = (const float*)d_in[3];
    const float* b1 = (const float*)d_in[4];
    const float* W2 = (const float*)d_in[5];
    const float* b2 = (const float*)d_in[6];
    float* out = (float*)d_out;

    char* ws = (char*)d_ws;
    float* probs = (float*)ws;                       // BT*2 f32 = 32 KB
    int*   cnt   = (int*)(ws + 64 * 1024);           // EE ints
    int*   list  = (int*)(ws + 64 * 1024 + 1024);    // EE*BT ints = 128 KB

    moe_zero_kernel<<<512, 256, 0, stream>>>(out, BT * DD, cnt);
    moe_router_kernel<<<BT / 8, 256, 0, stream>>>(x, Wr, br, probs, cnt, list);
    dim3 grid(BT / MT, EE);
    moe_mlp_kernel<<<grid, 256, 0, stream>>>(x, W1, b1, W2, b2, probs, cnt, list, out);
}
